// SAGPoolLayer_89859305767777
// MI455X (gfx1250) — matrix operand order repaired and checked by reading, not compiled, not run on hardware
//
#include <hip/hip_runtime.h>
#include <math.h>

// Problem constants (fixed by the reference)
#define N_NODES  200000
#define F_DIM    256
#define H_DIM    128
#define E_EDGES  3200000
#define G_GRAPHS 200
#define N_PER_G  1000
#define K_SEL    500
#define P_NODES  (G_GRAPHS * K_SEL)   // 100000
#define LN_EPS   1e-5f

typedef __attribute__((ext_vector_type(16))) __bf16 v16bf;
typedef __attribute__((ext_vector_type(8)))  float  v8f;

// ---------------------------------------------------------------------------
// Kernel 1: fused scoring pass.
//   h = x @ W1 + b1 ; LayerNorm ; exact GELU ; score = h @ W2 + b2
// GEMM runs on v_wmma_f32_16x16x32_bf16 with a 2-term bf16 split of both
// operands (3 WMMAs per tile) for ~fp32 accuracy, fp32 accumulate.
// Block = 128 threads (4 waves), each wave owns 16 rows; grid = N/64 = 3125.
// W1 is staged once per block into LDS, pre-swizzled into a contiguous-K B-operand
// fragment layout (hi and lo bf16 planes, 64 KB each -> 128 KB dynamic LDS).
// ---------------------------------------------------------------------------
__global__ void sag_scores_kernel(const float* __restrict__ x,
                                  const float* __restrict__ W1,
                                  const float* __restrict__ b1,
                                  const float* __restrict__ gamma,
                                  const float* __restrict__ beta,
                                  const float* __restrict__ W2,
                                  const float* __restrict__ b2,
                                  float* __restrict__ scores) {
  extern __shared__ __align__(16) char smem[];
  __bf16* sBhi = (__bf16*)smem;          // [8 ks][8 nt][32 lanes][16 halves]
  __bf16* sBlo = sBhi + 8 * 8 * 512;     // second 64 KB plane

  const int tid = threadIdx.x;

  // Cooperative fill: W1 (256x128 fp32, row-major) -> B-fragment order.
  // Read mapping (per 16x32-K chunk ks, 16-col tile nt):
  //   lane l holds col = nt*16 + (l&15), K = ks*32 + (l>>4)*16 + i, i=0..15.
  for (int idx = tid; idx < F_DIM * H_DIM; idx += blockDim.x) {
    const int k = idx >> 7;        // K index 0..255
    const int c = idx & 127;       // col   0..127
    const float f  = W1[idx];
    const __bf16 hi = (__bf16)f;
    const __bf16 lo = (__bf16)(f - (float)hi);
    const int ks   = k >> 5;
    const int kr   = k & 31;
    const int khh  = kr >> 4;
    const int i    = kr & 15;
    const int nt   = c >> 4;
    const int lane = (c & 15) | (khh << 4);
    const int off  = ((ks * 8 + nt) << 9) + (lane << 4) + i;
    sBhi[off] = hi;
    sBlo[off] = lo;
  }
  __syncthreads();

  const int l    = tid & 31;
  const int kh   = l >> 4;        // lane half (K split for A, row split for C)
  const int cl   = l & 15;
  const int wave = tid >> 5;
  const int r0   = (blockIdx.x * 4 + wave) * 16;   // 16-row tile base
  const int row  = r0 + cl;                        // A-matrix row of this lane

  // Per-lane column params (col = nt*16 + cl in C fragment)
  float b1v[8], gv[8], bev[8], w2v[8];
#pragma unroll
  for (int nt = 0; nt < 8; ++nt) {
    const int c = nt * 16 + cl;
    b1v[nt] = b1[c]; gv[nt] = gamma[c]; bev[nt] = beta[c]; w2v[nt] = W2[c];
  }
  const float b2s = b2[0];

  v8f acc[8] = {};   // 8 N-tiles of 16x16 fp32 accumulators

  const float* xrow = x + (size_t)row * F_DIM;

#pragma unroll
  for (int ks = 0; ks < 8; ++ks) {
    const float* p = xrow + ks * 32 + kh * 16;
    if (ks < 7) __builtin_prefetch(p + 32, 0, 0);   // global_prefetch_b8 next chunk
    // A fragment: halves 0..15 = K ks*32+kh*16+0..15
    const float4 f0 = *(const float4*)(p);
    const float4 f1 = *(const float4*)(p + 4);
    const float4 f2 = *(const float4*)(p + 8);
    const float4 f3 = *(const float4*)(p + 12);
    const float fv[16] = {f0.x, f0.y, f0.z, f0.w, f1.x, f1.y, f1.z, f1.w,
                          f2.x, f2.y, f2.z, f2.w, f3.x, f3.y, f3.z, f3.w};
    v16bf ahi, alo;
#pragma unroll
    for (int i = 0; i < 16; ++i) {
      const __bf16 h = (__bf16)fv[i];
      ahi[i] = h;
      alo[i] = (__bf16)(fv[i] - (float)h);
    }
#pragma unroll
    for (int nt = 0; nt < 8; ++nt) {
      const int cb = ((ks * 8 + nt) << 9) + (l << 4);   // 32B-aligned per lane
      const v16bf bhi = *(const v16bf*)(sBhi + cb);
      const v16bf blo = *(const v16bf*)(sBlo + cb);
      acc[nt] = __builtin_amdgcn_wmma_f32_16x16x32_bf16(
          false, ahi, false, bhi, (short)0, acc[nt], false, false);
      acc[nt] = __builtin_amdgcn_wmma_f32_16x16x32_bf16(
          false, alo, false, bhi, (short)0, acc[nt], false, false);
      acc[nt] = __builtin_amdgcn_wmma_f32_16x16x32_bf16(
          false, ahi, false, blo, (short)0, acc[nt], false, false);
    }
  }

  // + b1 (column bias: same for all j within a lane)
#pragma unroll
  for (int nt = 0; nt < 8; ++nt)
#pragma unroll
    for (int j = 0; j < 8; ++j)
      acc[nt][j] += b1v[nt];

  // LayerNorm over H=128 per row, then GELU(exact erf), then dot with W2.
  // C layout: acc[nt][j] @ lane l = h[r0 + j + 8*kh][nt*16 + cl].
  // Row reduction = sum over nt (in-lane) then butterfly over the 16-lane half.
  float p8[8];
#pragma unroll
  for (int j = 0; j < 8; ++j) {
    float s1 = 0.f, s2 = 0.f;
#pragma unroll
    for (int nt = 0; nt < 8; ++nt) { const float v = acc[nt][j]; s1 += v; s2 += v * v; }
#pragma unroll
    for (int m = 1; m <= 8; m <<= 1) {
      s1 += __shfl_xor(s1, m, 32);
      s2 += __shfl_xor(s2, m, 32);
    }
    const float mu  = s1 * (1.f / 128.f);
    const float var = s2 * (1.f / 128.f) - mu * mu;
    const float rs  = rsqrtf(var + LN_EPS);
    float pj = 0.f;
#pragma unroll
    for (int nt = 0; nt < 8; ++nt) {
      const float v = (acc[nt][j] - mu) * rs * gv[nt] + bev[nt];
      const float g = 0.5f * v * (1.f + erff(v * 0.70710678118654752f));
      pj += g * w2v[nt];
    }
#pragma unroll
    for (int m = 1; m <= 8; m <<= 1) pj += __shfl_xor(pj, m, 32);
    p8[j] = pj + b2s;
  }
  if (cl == 0) {   // lanes 0 and 16 write rows r0+0..7 and r0+8..15
#pragma unroll
    for (int j = 0; j < 8; ++j) scores[r0 + kh * 8 + j] = p8[j];
  }
}

// ---------------------------------------------------------------------------
// Kernel 2: exact segmented top-k by rank counting (matches lax.top_k order:
// descending, ties broken by lower index). rank<k => output slot = rank.
// One block (256 threads) per graph; 1000 scores live in LDS.
// Also emits node_map (-1 for dropped nodes), perm + batch_pooled outputs.
// ---------------------------------------------------------------------------
__global__ void sag_topk_kernel(const float* __restrict__ scores,
                                int* __restrict__ perm_idx,
                                int* __restrict__ node_map,
                                float* __restrict__ out_perm,
                                float* __restrict__ out_batch) {
  __shared__ float s[N_PER_G];
  const int g = blockIdx.x;
  const int tid = threadIdx.x;
  const float* sg = scores + (size_t)g * N_PER_G;
  for (int i = tid; i < N_PER_G; i += blockDim.x) s[i] = sg[i];
  __syncthreads();
  for (int i = tid; i < N_PER_G; i += blockDim.x) {
    const float si = s[i];
    int rank = 0;
    for (int j = 0; j < N_PER_G; ++j) {
      const float sj = s[j];
      if (sj > si || (sj == si && j < i)) ++rank;
    }
    const int node = g * N_PER_G + i;
    if (rank < K_SEL) {
      const int pos = g * K_SEL + rank;
      perm_idx[pos]  = node;
      node_map[node] = pos;
      out_perm[pos]  = (float)node;
      out_batch[pos] = (float)g;       // batch is sorted: batch[node]=g
    } else {
      node_map[node] = -1;
    }
  }
}

// ---------------------------------------------------------------------------
// Kernel 3: x_pooled = x[perm]  (one block per pooled row, float4 copies)
// ---------------------------------------------------------------------------
__global__ void sag_gather_kernel(const float* __restrict__ x,
                                  const int* __restrict__ perm_idx,
                                  float* __restrict__ out_x) {
  const int p = blockIdx.x;
  const int t = threadIdx.x;           // 64 threads * float4 = 256 floats
  const int src = perm_idx[p];
  const float4* xi = (const float4*)(x + (size_t)src * F_DIM);
  float4* xo = (float4*)(out_x + (size_t)p * F_DIM);
  xo[t] = xi[t];
}

// ---------------------------------------------------------------------------
// Kernel 4: edge mask + remap.
// out_ei[row, e] = mask ? node_map[edge_index[row, e]] : -1 ; mask as 0/1.
// ---------------------------------------------------------------------------
__global__ void sag_edge_kernel(const int* __restrict__ ei,
                                const int* __restrict__ node_map,
                                float* __restrict__ out_ei,
                                float* __restrict__ out_mask) {
  const int e = blockIdx.x * blockDim.x + threadIdx.x;
  if (e >= E_EDGES) return;
  const int sN = ei[e];
  const int dN = ei[E_EDGES + e];
  const int ms = node_map[sN];
  const int md = node_map[dN];
  const bool m = (ms >= 0) && (md >= 0);
  out_ei[e]            = m ? (float)ms : -1.0f;
  out_ei[E_EDGES + e]  = m ? (float)md : -1.0f;
  out_mask[e]          = m ? 1.0f : 0.0f;
}

// ---------------------------------------------------------------------------
extern "C" void kernel_launch(void* const* d_in, const int* in_sizes, int n_in,
                              void* d_out, int out_size, void* d_ws, size_t ws_size,
                              hipStream_t stream) {
  const float* x          = (const float*)d_in[0];
  const int*   edge_index = (const int*)  d_in[1];
  // d_in[2] = batch (implied by layout), d_in[3] = num_graphs (compile-time)
  const float* W1    = (const float*)d_in[4];
  const float* b1    = (const float*)d_in[5];
  const float* gamma = (const float*)d_in[6];
  const float* beta  = (const float*)d_in[7];
  const float* W2    = (const float*)d_in[8];
  const float* b2    = (const float*)d_in[9];

  // Outputs concatenated flat (float) in reference return order.
  float* out       = (float*)d_out;
  float* out_x     = out;                                       // P*F
  float* out_ei    = out_x  + (size_t)P_NODES * F_DIM;          // 2*E
  float* out_batch = out_ei + 2 * (size_t)E_EDGES;              // P
  float* out_perm  = out_batch + P_NODES;                       // P
  float* out_mask  = out_perm  + P_NODES;                       // E

  // Workspace: scores[N] f32, perm_idx[P] i32, node_map[N] i32  (~2.0 MB)
  char*  ws       = (char*)d_ws;
  float* scores   = (float*)ws;
  int*   perm_idx = (int*)(ws + (size_t)N_NODES * 4);
  int*   node_map = (int*)(ws + (size_t)N_NODES * 4 + (size_t)P_NODES * 4);

  // 1) fused scoring (WMMA): 3125 blocks x 128 thr, 128 KB dynamic LDS
  sag_scores_kernel<<<N_NODES / 64, 128, 2 * 8 * 8 * 512 * (int)sizeof(__bf16),
                      stream>>>(x, W1, b1, gamma, beta, W2, b2, scores);

  // 2) segmented top-k: one block per graph
  sag_topk_kernel<<<G_GRAPHS, 256, 0, stream>>>(scores, perm_idx, node_map,
                                                out_perm, out_batch);

  // 3) x gather
  sag_gather_kernel<<<P_NODES, 64, 0, stream>>>(x, perm_idx, out_x);

  // 4) edges
  sag_edge_kernel<<<(E_EDGES + 255) / 256, 256, 0, stream>>>(edge_index, node_map,
                                                             out_ei, out_mask);
}
